// TransformerBlock_44942537785623
// MI455X (gfx1250) — compile-verified, matrix operand order checked by reading, not run on hardware
//
#include <hip/hip_runtime.h>

typedef unsigned short u16;
typedef __attribute__((ext_vector_type(16))) __bf16 bf16x16;
typedef __attribute__((ext_vector_type(8)))  float  f32x8;

union FragU { bf16x16 v; uint4 q[2]; };

__device__ __forceinline__ u16 f2bf(float f) {
    unsigned u = __float_as_uint(f);
    u += 0x7FFFu + ((u >> 16) & 1u);   // round-to-nearest-even
    return (u16)(u >> 16);
}

__device__ __forceinline__ f32x8 zero8() {
    f32x8 z;
#pragma unroll
    for (int i = 0; i < 8; ++i) z[i] = 0.0f;
    return z;
}

// Load one 16x32 bf16 operand fragment (A-layout; B uses transposed source so
// the same layout applies). Lanes 0-15: rows 0-15, K 0-7 & 16-23.
// Lanes 16-31: rows 0-15, K 8-15 & 24-31. Two 16B loads per lane.
__device__ __forceinline__ bf16x16 load_frag(const u16* base, int stride, int lane) {
    const int half = lane >> 4, r = lane & 15;
    const u16* p = base + (size_t)r * stride + half * 8;
    FragU u;
    u.q[0] = *(const uint4*)(p);
    u.q[1] = *(const uint4*)(p + 16);
    return u.v;
}

__device__ __forceinline__ f32x8 wmma32(bf16x16 a, bf16x16 b, f32x8 c) {
    return __builtin_amdgcn_wmma_f32_16x16x32_bf16(false, a, false, b, (short)0, c, false, false);
}

__device__ __forceinline__ float gelu_exact(float x) {
    return 0.5f * x * (1.0f + erff(x * 0.70710678118654752f));
}

// ---------------------------------------------------------------------------
// Weight cast + transpose: in (K x N) f32 row-major -> out (N x K) bf16
// ---------------------------------------------------------------------------
__global__ __launch_bounds__(256) void cast_transpose_k(
    const float* __restrict__ in, u16* __restrict__ out, int K, int N) {
    __shared__ float tile[32][33];
    const int n0 = blockIdx.x * 32, k0 = blockIdx.y * 32;
    const int tx = threadIdx.x, ty = threadIdx.y;
#pragma unroll
    for (int i = 0; i < 32; i += 8)
        tile[ty + i][tx] = in[(size_t)(k0 + ty + i) * N + (n0 + tx)];
    __syncthreads();
#pragma unroll
    for (int i = 0; i < 32; i += 8)
        out[(size_t)(n0 + ty + i) * K + (k0 + tx)] = f2bf(tile[tx][ty + i]);
}

// ---------------------------------------------------------------------------
// LayerNorm (f32 in, bf16 out), one 256-thread block per row of D=1024
// ---------------------------------------------------------------------------
__global__ __launch_bounds__(256) void layernorm_k(
    const float* __restrict__ in, const float* __restrict__ g,
    const float* __restrict__ be, u16* __restrict__ out) {
    const int row = blockIdx.x, tid = threadIdx.x;
    const float* rp = in + (size_t)row * 1024;
    float v[4], s = 0.0f, ss = 0.0f;
#pragma unroll
    for (int i = 0; i < 4; ++i) {
        v[i] = rp[tid + 256 * i];
        s += v[i];
        ss += v[i] * v[i];
    }
    __shared__ float rs[256], rq[256];
    rs[tid] = s; rq[tid] = ss;
    __syncthreads();
    for (int off = 128; off > 0; off >>= 1) {
        if (tid < off) { rs[tid] += rs[tid + off]; rq[tid] += rq[tid + off]; }
        __syncthreads();
    }
    const float mu = rs[0] * (1.0f / 1024.0f);
    const float var = rq[0] * (1.0f / 1024.0f) - mu * mu;
    const float rstd = rsqrtf(var + 1e-5f);
    u16* op = out + (size_t)row * 1024;
#pragma unroll
    for (int i = 0; i < 4; ++i) {
        const int col = tid + 256 * i;
        op[col] = f2bf((v[i] - mu) * rstd * g[col] + be[col]);
    }
}

// ---------------------------------------------------------------------------
// WMMA GEMM: C[M,N] = act(A[M,K] @ B[K,N] + bias) (+ residual)
// A: bf16 row-major, BT: bf16 N x K (transposed), bias f32.
// 256 threads = 8 waves; block tile 128x128; wave tile 32(M)x64(N).
// ---------------------------------------------------------------------------
template <int OUT_BF16, int ACT_GELU, int ADD_RES>
__global__ __launch_bounds__(256) void gemm_k(
    const u16* __restrict__ A, const u16* __restrict__ BT,
    const float* __restrict__ bias, const float* __restrict__ res,
    void* __restrict__ outp, int M, int N, int K) {
    const int tid = threadIdx.x, lane = tid & 31, wave = tid >> 5;
    const int wm = wave & 3, wn = wave >> 2;
    const int m0 = blockIdx.y * 128 + wm * 32;
    const int n0 = blockIdx.x * 128 + wn * 64;

    f32x8 acc[2][4];
#pragma unroll
    for (int i = 0; i < 2; ++i)
#pragma unroll
        for (int j = 0; j < 4; ++j) acc[i][j] = zero8();

    for (int k0 = 0; k0 < K; k0 += 32) {
        bf16x16 af[2], bf[4];
#pragma unroll
        for (int i = 0; i < 2; ++i)
            af[i] = load_frag(A + (size_t)(m0 + i * 16) * K + k0, K, lane);
#pragma unroll
        for (int j = 0; j < 4; ++j)
            bf[j] = load_frag(BT + (size_t)(n0 + j * 16) * K + k0, K, lane);
#pragma unroll
        for (int i = 0; i < 2; ++i)
#pragma unroll
            for (int j = 0; j < 4; ++j)
                acc[i][j] = wmma32(af[i], bf[j], acc[i][j]);
    }

    const int c = lane & 15, hs = lane >> 4;
#pragma unroll
    for (int i = 0; i < 2; ++i) {
#pragma unroll
        for (int j = 0; j < 4; ++j) {
            const int col = n0 + j * 16 + c;
            const float bv = bias[col];
#pragma unroll
            for (int v = 0; v < 8; ++v) {
                const int row = m0 + i * 16 + v + 8 * hs;
                float val = acc[i][j][v] + bv;
                if (ACT_GELU) val = gelu_exact(val);
                if (ADD_RES) val += res[(size_t)row * N + col];
                if (OUT_BF16)
                    ((u16*)outp)[(size_t)row * N + col] = f2bf(val);
                else
                    ((float*)outp)[(size_t)row * N + col] = val;
            }
        }
    }
}

// ---------------------------------------------------------------------------
// V transpose per (b,h): V (b,s,h,d) bf16 -> VT (b*h, d, s) bf16
// ---------------------------------------------------------------------------
__global__ __launch_bounds__(256) void transpose_v_k(
    const u16* __restrict__ V, u16* __restrict__ VT, int S) {
    __shared__ u16 tile[32][33];
    const int s0 = blockIdx.x * 32, d0 = blockIdx.y * 32, bh = blockIdx.z;
    const int b = bh >> 4, h = bh & 15;
    const int tx = threadIdx.x, ty = threadIdx.y;
#pragma unroll
    for (int i = 0; i < 32; i += 8)
        tile[ty + i][tx] = V[(size_t)(b * S + s0 + ty + i) * 1024 + h * 64 + d0 + tx];
    __syncthreads();
#pragma unroll
    for (int i = 0; i < 32; i += 8)
        VT[(size_t)(bh * 64 + d0 + ty + i) * S + (s0 + tx)] = tile[tx][ty + i];
}

// ---------------------------------------------------------------------------
// Causal flash attention, dk = 64. Grid: (S/128, B*H), 8 waves per block,
// each wave owns 16 query rows; streams 32-key blocks with online softmax.
// ---------------------------------------------------------------------------
__global__ __launch_bounds__(256) void attn_k(
    const u16* __restrict__ Q, const u16* __restrict__ Kb,
    const u16* __restrict__ VT, u16* __restrict__ CTX, int S) {
    const int tid = threadIdx.x, lane = tid & 31, wave = tid >> 5;
    const int bh = blockIdx.y, b = bh >> 4, h = bh & 15;
    const int q0 = blockIdx.x * 128 + wave * 16;
    const int c = lane & 15, hs = lane >> 4;

    __shared__ __align__(16) u16 pbuf[8][640];   // 16 rows x stride 40 per wave
    u16* pw = pbuf[wave];

    const u16* qbase = Q + (size_t)(b * S + q0) * 1024 + h * 64;
    const bf16x16 aq0 = load_frag(qbase, 1024, lane);        // dk 0..31
    const bf16x16 aq1 = load_frag(qbase + 32, 1024, lane);   // dk 32..63

    float mrow[8], lsum[8];
    f32x8 cacc[4];
#pragma unroll
    for (int v = 0; v < 8; ++v) { mrow[v] = -1e30f; lsum[v] = 0.0f; }
#pragma unroll
    for (int n = 0; n < 4; ++n) cacc[n] = zero8();

    const float SC = 0.125f;            // 1/sqrt(64)
    const float L2E = 1.44269504f;

    for (int kb = 0; kb < q0 + 16; kb += 32) {
        // ---- scores: S = Q(16x64) @ K^T(64x32) in two 16-key column frags
        f32x8 s0 = zero8(), s1 = zero8();
        const u16* kp = Kb + (size_t)(b * S + kb) * 1024 + h * 64;
        s0 = wmma32(aq0, load_frag(kp, 1024, lane), s0);
        s1 = wmma32(aq0, load_frag(kp + (size_t)16 * 1024, 1024, lane), s1);
        s0 = wmma32(aq1, load_frag(kp + 32, 1024, lane), s0);
        s1 = wmma32(aq1, load_frag(kp + (size_t)16 * 1024 + 32, 1024, lane), s1);

        // ---- online softmax per row (rows live in (vgpr, lane-half))
#pragma unroll
        for (int v = 0; v < 8; ++v) {
            const int row = q0 + v + 8 * hs;
            const int key0 = kb + c, key1 = key0 + 16;
            float f0 = (key0 <= row) ? s0[v] * SC : -1e30f;
            float f1 = (key1 <= row) ? s1[v] * SC : -1e30f;
            float mx = fmaxf(f0, f1);
#pragma unroll
            for (int off = 8; off; off >>= 1) mx = fmaxf(mx, __shfl_xor(mx, off, 32));
            const float mnew = fmaxf(mrow[v], mx);
            const float p0 = exp2f((f0 - mnew) * L2E);
            const float p1 = exp2f((f1 - mnew) * L2E);
            float ps = p0 + p1;
#pragma unroll
            for (int off = 8; off; off >>= 1) ps += __shfl_xor(ps, off, 32);
            const float alpha = exp2f((mrow[v] - mnew) * L2E);
            lsum[v] = lsum[v] * alpha + ps;
            mrow[v] = mnew;
#pragma unroll
            for (int n = 0; n < 4; ++n) cacc[n][v] *= alpha;
            const int prow = v + 8 * hs;
            pw[prow * 40 + c] = f2bf(p0);
            pw[prow * 40 + 16 + c] = f2bf(p1);
        }
        asm volatile("s_wait_dscnt 0" ::: "memory");   // LDS stores visible to wave
        const bf16x16 pfrag = load_frag(pw, 40, lane); // P (16x32) in A layout

        // ---- ctx += P(16x32) @ V(32x64), VT gives contiguous B frags
        const u16* vtb = VT + (size_t)(bh * 64) * S + kb;
#pragma unroll
        for (int n = 0; n < 4; ++n)
            cacc[n] = wmma32(pfrag, load_frag(vtb + (size_t)(n * 16) * S, S, lane), cacc[n]);
    }

    float inv[8];
#pragma unroll
    for (int v = 0; v < 8; ++v) inv[v] = 1.0f / lsum[v];
#pragma unroll
    for (int n = 0; n < 4; ++n)
#pragma unroll
        for (int v = 0; v < 8; ++v) {
            const int row = q0 + v + 8 * hs;
            const int col = n * 16 + c;
            CTX[(size_t)(b * S + row) * 1024 + h * 64 + col] = f2bf(cacc[n][v] * inv[v]);
        }
}

// ---------------------------------------------------------------------------
// Host-side orchestration
// ---------------------------------------------------------------------------
extern "C" void kernel_launch(void* const* d_in, const int* in_sizes, int n_in,
                              void* d_out, int out_size, void* d_ws, size_t ws_size,
                              hipStream_t stream) {
    constexpr int B = 4, S = 2048, D = 1024, H = 16, F = 4096;
    constexpr int M = B * S;  // 8192 tokens
    constexpr size_t MiB = 1u << 20;

    const float* x    = (const float*)d_in[0];
    const float* Wq   = (const float*)d_in[1];
    const float* bq   = (const float*)d_in[2];
    const float* Wk   = (const float*)d_in[3];
    const float* bk   = (const float*)d_in[4];
    const float* Wv   = (const float*)d_in[5];
    const float* bv   = (const float*)d_in[6];
    const float* Wo   = (const float*)d_in[7];
    const float* bo   = (const float*)d_in[8];
    const float* ln1g = (const float*)d_in[9];
    const float* ln1b = (const float*)d_in[10];
    const float* ln2g = (const float*)d_in[11];
    const float* ln2b = (const float*)d_in[12];
    const float* W1   = (const float*)d_in[13];
    const float* b1   = (const float*)d_in[14];
    const float* W2   = (const float*)d_in[15];
    const float* b2   = (const float*)d_in[16];

    char* ws = (char*)d_ws;
    u16*   WQT = (u16*)(ws + 0 * MiB);     // D x D bf16 (transposed)  2 MiB
    u16*   WKT = (u16*)(ws + 2 * MiB);
    u16*   WVT = (u16*)(ws + 4 * MiB);
    u16*   WOT = (u16*)(ws + 6 * MiB);
    u16*   W1T = (u16*)(ws + 8 * MiB);     // F x D bf16               8 MiB
    u16*   W2T = (u16*)(ws + 16 * MiB);    // D x F bf16               8 MiB
    u16*   H1  = (u16*)(ws + 24 * MiB);    // M x D bf16              16 MiB
    u16*   QB  = (u16*)(ws + 40 * MiB);    // M x D bf16
    u16*   KB  = (u16*)(ws + 56 * MiB);
    u16*   VB  = (u16*)(ws + 72 * MiB);
    u16*   VTB = (u16*)(ws + 88 * MiB);    // (B*H) x 64 x S bf16
    u16*   CTX = (u16*)(ws + 104 * MiB);   // M x D bf16
    float* X1  = (float*)(ws + 120 * MiB); // M x D f32               32 MiB
    u16*   U   = (u16*)(ws + 40 * MiB);    // M x F bf16 (reuses Q/K/V/VT) 64 MiB
    float* OUT = (float*)d_out;
    (void)in_sizes; (void)n_in; (void)out_size; (void)ws_size;

    const dim3 tb(32, 8);
    // 0) weights -> bf16 transposed (L2-resident working set: 24 MiB)
    cast_transpose_k<<<dim3(D / 32, D / 32), tb, 0, stream>>>(Wq, WQT, D, D);
    cast_transpose_k<<<dim3(D / 32, D / 32), tb, 0, stream>>>(Wk, WKT, D, D);
    cast_transpose_k<<<dim3(D / 32, D / 32), tb, 0, stream>>>(Wv, WVT, D, D);
    cast_transpose_k<<<dim3(D / 32, D / 32), tb, 0, stream>>>(Wo, WOT, D, D);
    cast_transpose_k<<<dim3(F / 32, D / 32), tb, 0, stream>>>(W1, W1T, D, F);
    cast_transpose_k<<<dim3(D / 32, F / 32), tb, 0, stream>>>(W2, W2T, F, D);

    // 1) LN1: x -> H1 (bf16)
    layernorm_k<<<M, 256, 0, stream>>>(x, ln1g, ln1b, H1);

    // 2) Q/K/V projections (bf16 out)
    gemm_k<1, 0, 0><<<dim3(D / 128, M / 128), 256, 0, stream>>>(H1, WQT, bq, nullptr, QB, M, D, D);
    gemm_k<1, 0, 0><<<dim3(D / 128, M / 128), 256, 0, stream>>>(H1, WKT, bk, nullptr, KB, M, D, D);
    gemm_k<1, 0, 0><<<dim3(D / 128, M / 128), 256, 0, stream>>>(H1, WVT, bv, nullptr, VB, M, D, D);

    // 3) V -> VT per (b,h)
    transpose_v_k<<<dim3(S / 32, 2, B * H), tb, 0, stream>>>(VB, VTB, S);

    // 4) causal flash attention -> CTX (bf16)
    attn_k<<<dim3(S / 128, B * H), 256, 0, stream>>>(QB, KB, VTB, CTX, S);

    // 5) output projection + residual: X1 = x + CTX @ Wo + bo (f32)
    gemm_k<0, 0, 1><<<dim3(D / 128, M / 128), 256, 0, stream>>>(CTX, WOT, bo, x, X1, M, D, D);

    // 6) LN2: X1 -> H1 (bf16, reuse)
    layernorm_k<<<M, 256, 0, stream>>>(X1, ln2g, ln2b, H1);

    // 7) FFN up + exact GELU -> U (bf16)
    gemm_k<1, 1, 0><<<dim3(F / 128, M / 128), 256, 0, stream>>>(H1, W1T, b1, nullptr, U, M, F, D);

    // 8) FFN down + residual -> OUT (f32)
    gemm_k<0, 0, 1><<<dim3(D / 128, M / 128), 256, 0, stream>>>(U, W2T, b2, X1, OUT, M, D, F);
}
